// GptOssGate_76656576299581
// MI455X (gfx1250) — matrix operand order repaired and checked by reading, not compiled, not run on hardware
//
#include <hip/hip_runtime.h>

// ---------------- problem constants (match reference) ----------------
#define H_DIM     2880
#define E_DIM     32
#define TOPK      4
#define KSTEP     32                    // K per v_wmma_f32_16x16x32_bf16
#define NK        (H_DIM / KSTEP)       // 90 k-steps
#define NK_HALF   (NK / 2)              // 45 per H-split half
#define WAVE_TOK  32                    // tokens per wave (two 16-row A tiles)
#define TOK_GROUPS 4
#define BLOCK_TOK (WAVE_TOK * TOK_GROUPS)   // 128 tokens per workgroup
#define BLOCK_THREADS 256                   // 8 waves: 4 token groups x 2 H halves
#define LDS_STRIDE (E_DIM + 1)              // pad to dodge bank conflicts

typedef __attribute__((ext_vector_type(16))) __bf16 v16bf;
typedef __attribute__((ext_vector_type(8)))  float  v8f;

__device__ __forceinline__ v16bf pack16(float4 f0, float4 f1, float4 f2, float4 f3) {
  v16bf r;
  r[0]=(__bf16)f0.x;  r[1]=(__bf16)f0.y;  r[2]=(__bf16)f0.z;  r[3]=(__bf16)f0.w;
  r[4]=(__bf16)f1.x;  r[5]=(__bf16)f1.y;  r[6]=(__bf16)f1.z;  r[7]=(__bf16)f1.w;
  r[8]=(__bf16)f2.x;  r[9]=(__bf16)f2.y;  r[10]=(__bf16)f2.z; r[11]=(__bf16)f2.w;
  r[12]=(__bf16)f3.x; r[13]=(__bf16)f3.y; r[14]=(__bf16)f3.z; r[15]=(__bf16)f3.w;
  return r;
}

// A fragment (16-bit A 16x32): lane holds row lane%16,
// K = {kb..kb+15}, kb = (lane>>4)*16. p already includes kb.
__device__ __forceinline__ v16bf load_a_frag(const float* __restrict__ p) {
  const float4* q = (const float4*)p;
  return pack16(q[0], q[1], q[2], q[3]);   // floats 0..15
}
// B fragment (K x 16): lane holds col lane%16, contiguous K block of 16
// starting at (lane>>4)*16. p already includes that sub-base.
__device__ __forceinline__ v16bf load_b_frag(const float* __restrict__ p) {
  const float4* q = (const float4*)p;
  return pack16(q[0], q[1], q[2], q[3]);   // floats 0..15
}

__global__ __launch_bounds__(BLOCK_THREADS)
void moe_router_topk_kernel(const float* __restrict__ hidden,
                            const float* __restrict__ weight,
                            const float* __restrict__ bias,
                            float* __restrict__ out, int T)
{
  __shared__ float part[2][BLOCK_TOK][LDS_STRIDE];  // two H-halves of logits

  const int tid  = threadIdx.x;
  const int wave = tid >> 5;
  const int lane = tid & 31;
  const int m    = lane & 15;
  const int hi   = lane >> 4;       // 0/1: which half of the lane pattern
  const int g    = wave & 3;        // token group within block
  const int hh   = wave >> 2;       // H half (0: k<1440, 1: k>=1440)

  const int tokBase = blockIdx.x * BLOCK_TOK + g * WAVE_TOK;

  int r0 = tokBase + m;       if (r0 > T - 1) r0 = T - 1;
  int r1 = tokBase + 16 + m;  if (r1 > T - 1) r1 = T - 1;

  const int kOff = hh * (NK_HALF * KSTEP);
  const float* aRow0 = hidden + (size_t)r0 * H_DIM + kOff + hi * 16;
  const float* aRow1 = hidden + (size_t)r1 * H_DIM + kOff + hi * 16;
  const float* bRow0 = weight + (size_t)(m)      * H_DIM + kOff + hi * 16;
  const float* bRow1 = weight + (size_t)(16 + m) * H_DIM + kOff + hi * 16;

  v8f acc00 = {}; v8f acc01 = {}; v8f acc10 = {}; v8f acc11 = {};

  #pragma unroll 3
  for (int ks = 0; ks < NK_HALF; ++ks) {
    const int k0 = ks * KSTEP;
    v16bf b0 = load_b_frag(bRow0 + k0);   // experts 0..15  (L2-resident)
    v16bf b1 = load_b_frag(bRow1 + k0);   // experts 16..31
    v16bf a0 = load_a_frag(aRow0 + k0);   // tokens tokBase..+15 (HBM stream)
    v16bf a1 = load_a_frag(aRow1 + k0);   // tokens tokBase+16..+31
    acc00 = __builtin_amdgcn_wmma_f32_16x16x32_bf16(false, a0, false, b0, (short)0, acc00, false, false);
    acc01 = __builtin_amdgcn_wmma_f32_16x16x32_bf16(false, a0, false, b1, (short)0, acc01, false, false);
    acc10 = __builtin_amdgcn_wmma_f32_16x16x32_bf16(false, a1, false, b0, (short)0, acc10, false, false);
    acc11 = __builtin_amdgcn_wmma_f32_16x16x32_bf16(false, a1, false, b1, (short)0, acc11, false, false);
  }

  // C/D layout: lane L, VGPR j -> M = j + 8*(L>>4), N = L%16
  const int tg = g * WAVE_TOK;
  #pragma unroll
  for (int j = 0; j < 8; ++j) {
    const int row0 = tg + j + 8 * hi;       // token-local row for first tile
    const int row1 = row0 + 16;             // second tile
    part[hh][row0][m]      = acc00[j];
    part[hh][row0][16 + m] = acc01[j];
    part[hh][row1][m]      = acc10[j];
    part[hh][row1][16 + m] = acc11[j];
  }
  __syncthreads();

  // ---- per-token: reduce halves, bias, top-4 (sorted desc), softmax ----
  if (tid < BLOCK_TOK) {
    const int t = blockIdx.x * BLOCK_TOK + tid;
    if (t < T) {
      const float NEG = -3.402823466e38f;
      float v0 = NEG, v1 = NEG, v2 = NEG, v3 = NEG;
      int   i0 = 0,  i1 = 0,  i2 = 0,  i3 = 0;
      #pragma unroll
      for (int e = 0; e < E_DIM; ++e) {
        float x = part[0][tid][e] + part[1][tid][e] + bias[e];
        if (x > v0)      { v3=v2;i3=i2; v2=v1;i2=i1; v1=v0;i1=i0; v0=x; i0=e; }
        else if (x > v1) { v3=v2;i3=i2; v2=v1;i2=i1; v1=x; i1=e; }
        else if (x > v2) { v3=v2;i3=i2; v2=x; i2=e; }
        else if (x > v3) { v3=x; i3=e; }
      }
      // softmax over the 4 selected logits (v0 is the max)
      float w0 = 1.0f;                 // expf(v0 - v0)
      float w1 = __expf(v1 - v0);
      float w2 = __expf(v2 - v0);
      float w3 = __expf(v3 - v0);
      float inv = 1.0f / (w0 + w1 + w2 + w3);

      float* oidx = out;                        // topk_idx  [T,4] (as float)
      float* ow   = out + (size_t)T * TOPK;     // topk_weight [T,4]
      const size_t b = (size_t)t * TOPK;
      oidx[b + 0] = (float)i0; oidx[b + 1] = (float)i1;
      oidx[b + 2] = (float)i2; oidx[b + 3] = (float)i3;
      ow[b + 0] = w0 * inv; ow[b + 1] = w1 * inv;
      ow[b + 2] = w2 * inv; ow[b + 3] = w3 * inv;
    }
  }
}

extern "C" void kernel_launch(void* const* d_in, const int* in_sizes, int n_in,
                              void* d_out, int out_size, void* d_ws, size_t ws_size,
                              hipStream_t stream) {
  const float* hidden = (const float*)d_in[0];   // [B,S,H] f32
  const float* weight = (const float*)d_in[1];   // [E,H]   f32
  const float* bias   = (const float*)d_in[2];   // [E]     f32
  float* out = (float*)d_out;                    // idx[T,4] ++ weight[T,4]

  const int T = in_sizes[0] / H_DIM;             // 16384 tokens
  const int grid = (T + BLOCK_TOK - 1) / BLOCK_TOK;  // 128 workgroups

  hipLaunchKernelGGL(moe_router_topk_kernel, dim3(grid), dim3(BLOCK_THREADS), 0,
                     stream, hidden, weight, bias, out, T);
}
